// GCN_25159918420461
// MI455X (gfx1250) — matrix operand order repaired and checked by reading, not compiled, not run on hardware
//
#include <hip/hip_runtime.h>
#include <hip/hip_bf16.h>

// ---------------------------------------------------------------------------
// GCN forward for gfx1250 (MI455X).
//   - edge aggregation: fp32 global atomics (L2-resident, 23.3TB/s-class path)
//   - dense layers: v_wmma_f32_16x16x32_bf16, f32 accumulate
// ---------------------------------------------------------------------------

typedef __bf16 bf16;
typedef bf16  v16bf __attribute__((ext_vector_type(16)));
typedef bf16  v8bf  __attribute__((ext_vector_type(8)));
typedef float v8f   __attribute__((ext_vector_type(8)));

#define FEAT 256   // F == H == 256
#define MLPH 64
#define NCLS 2

// ---------------------------------------------------------------------------
// 1) degree accumulation (atomic +1.0 per edge endpoint)
// ---------------------------------------------------------------------------
__global__ void gcn_deg_kernel(const int* __restrict__ src, const int* __restrict__ dst,
                               float* __restrict__ deg_out, float* __restrict__ deg_in, int E) {
    int e = blockIdx.x * blockDim.x + threadIdx.x;
    if (e >= E) return;
    unsafeAtomicAdd(&deg_out[src[e]], 1.0f);
    unsafeAtomicAdd(&deg_in[dst[e]], 1.0f);
}

// deg -> rsqrt(max(deg,1)) in place, both arrays at once
__global__ void gcn_norm_kernel(float* __restrict__ a, float* __restrict__ b, int n) {
    int i = blockIdx.x * blockDim.x + threadIdx.x;
    if (i >= n) return;
    float da = a[i]; da = da < 1.0f ? 1.0f : da; a[i] = rsqrtf(da);
    float db = b[i]; db = db < 1.0f ? 1.0f : db; b[i] = rsqrtf(db);
}

// ---------------------------------------------------------------------------
// 2) per-node scale: out[i][:] = in[i][:] * norm[i]   (float4 vectorized)
// ---------------------------------------------------------------------------
__global__ void gcn_scale_kernel(const float* __restrict__ in, const float* __restrict__ norm,
                                 float* __restrict__ out, int n_nodes) {
    int t = blockIdx.x * blockDim.x + threadIdx.x;
    if (t >= n_nodes * (FEAT / 4)) return;
    int node = t >> 6;               // 64 float4 chunks per node
    int c = (t & 63) << 2;
    float s = norm[node];
    const float4 v = *(const float4*)(in + (size_t)node * FEAT + c);
    float4 o; o.x = v.x * s; o.y = v.y * s; o.z = v.z * s; o.w = v.w * s;
    *(float4*)(out + (size_t)node * FEAT + c) = o;
}

// ---------------------------------------------------------------------------
// 3) edge aggregation: agg[dst] += h[src]  (64 threads / edge, float4 gather,
//    4x global_atomic_add_f32; both h and agg are L2-resident at 51MB each)
// ---------------------------------------------------------------------------
__global__ void gcn_agg_kernel(const float* __restrict__ h, const int* __restrict__ src,
                               const int* __restrict__ dst, float* __restrict__ agg, int E) {
    int t = blockIdx.x * blockDim.x + threadIdx.x;
    int e = t >> 6;
    if (e >= E) return;
    int c = (t & 63) << 2;
    int s = src[e], d = dst[e];
    const float4 v = *(const float4*)(h + (size_t)s * FEAT + c);
    float* o = agg + (size_t)d * FEAT + c;
    unsafeAtomicAdd(o + 0, v.x);
    unsafeAtomicAdd(o + 1, v.y);
    unsafeAtomicAdd(o + 2, v.z);
    unsafeAtomicAdd(o + 3, v.w);
}

// ---------------------------------------------------------------------------
// 4) convert f32 row-features -> bf16 (optionally scaled by per-node norm)
// ---------------------------------------------------------------------------
__global__ void gcn_tobf16_kernel(const float* __restrict__ in, const float* __restrict__ norm,
                                  bf16* __restrict__ out, int n_nodes) {
    int t = blockIdx.x * blockDim.x + threadIdx.x;
    if (t >= n_nodes * (FEAT / 4)) return;
    int node = t >> 6;
    int c = (t & 63) << 2;
    float s = norm ? norm[node] : 1.0f;
    const float4 v = *(const float4*)(in + (size_t)node * FEAT + c);
    bf16* o = out + (size_t)node * FEAT + c;
    o[0] = (bf16)(v.x * s); o[1] = (bf16)(v.y * s);
    o[2] = (bf16)(v.z * s); o[3] = (bf16)(v.w * s);
}

// ---------------------------------------------------------------------------
// 5) weight prep: W[K][HO] f32 -> WT[HO][K] bf16 (so B-fragment loads are
//    two contiguous 16B runs per lane)
// ---------------------------------------------------------------------------
__global__ void gcn_wprep_kernel(const float* __restrict__ W, bf16* __restrict__ WT,
                                 int Kdim, int HOdim) {
    int t = blockIdx.x * blockDim.x + threadIdx.x;
    if (t >= Kdim * HOdim) return;
    int n = t / Kdim, k = t % Kdim;            // write coalesced over k
    WT[(size_t)n * Kdim + k] = (bf16)W[(size_t)k * HOdim + n];
}

// ---------------------------------------------------------------------------
// 6) WMMA GEMM: C[nrows x HO] = A[nrows x 256](bf16) @ WT^T + bias, opt. ReLU
//    8 waves / block; each wave owns one 16x16 tile, K=256 in 8 WMMA steps.
//    A-frag layout: lane(m=l&15,hi=l>>4): elems 0-7 <- K=k0+hi*8..+7,
//                                         elems 8-15 <- K=k0+16+hi*8..+7
//    B-frag layout: lane l: col n=l&15, same K order as A-frag (from WT rows)
//    C/D layout:    VGPR r -> row m_base+hi*8+r, lane -> col
// ---------------------------------------------------------------------------
template <int HO>
__global__ __launch_bounds__(256) void gcn_gemm_wmma(const bf16* __restrict__ A,
                                                     const bf16* __restrict__ WT,
                                                     const float* __restrict__ bias,
                                                     float* __restrict__ C,
                                                     int nrows, int do_relu) {
    const int wave = threadIdx.x >> 5;
    const int lane = threadIdx.x & 31;
    const int m_base = (blockIdx.x * 8 + wave) * 16;
    if (m_base >= nrows) return;                 // wave-uniform: EXEC stays all-1
    const int n_base = blockIdx.y * 16;
    const int m = lane & 15;
    const int hi = lane >> 4;
    const int K = FEAT;

    const bf16* Arow = A  + (size_t)(m_base + m) * K;
    const bf16* Brow = WT + (size_t)(n_base + m) * K;

    v8f acc = {};
#pragma unroll
    for (int k0 = 0; k0 < K; k0 += 32) {
        union { v16bf v; v8bf h[2]; } a;
        a.h[0] = *(const v8bf*)(Arow + k0 + hi * 8);
        a.h[1] = *(const v8bf*)(Arow + k0 + 16 + hi * 8);
        union { v16bf v; v8bf h[2]; } bu;
        bu.h[0] = *(const v8bf*)(Brow + k0 + hi * 8);
        bu.h[1] = *(const v8bf*)(Brow + k0 + 16 + hi * 8);
        v16bf b = bu.v;
        acc = __builtin_amdgcn_wmma_f32_16x16x32_bf16(
                  false, a.v, false, b, (short)0, acc, false, false);
    }

    const int n = n_base + m;
    const float bv = bias[n];
#pragma unroll
    for (int r = 0; r < 8; r++) {
        float v = acc[r] + bv;
        if (do_relu) v = fmaxf(v, 0.0f);
        C[(size_t)(m_base + hi * 8 + r) * HO + n] = v;
    }
}

// ---------------------------------------------------------------------------
// 7) final tiny layer: out[n][0:2] = hm[n][0:64] @ Wm2[64][2] + bm2
// ---------------------------------------------------------------------------
__global__ void gcn_mlp2_kernel(const float* __restrict__ hm, const float* __restrict__ Wm2,
                                const float* __restrict__ bm2, float* __restrict__ out, int n) {
    int i = blockIdx.x * blockDim.x + threadIdx.x;
    if (i >= n) return;
    float a0 = bm2[0], a1 = bm2[1];
    const float* h = hm + (size_t)i * MLPH;
#pragma unroll
    for (int k = 0; k < MLPH; k++) {
        float v = h[k];
        a0 += v * Wm2[k * NCLS + 0];
        a1 += v * Wm2[k * NCLS + 1];
    }
    out[(size_t)i * NCLS + 0] = a0;
    out[(size_t)i * NCLS + 1] = a1;
}

// ---------------------------------------------------------------------------
extern "C" void kernel_launch(void* const* d_in, const int* in_sizes, int n_in,
                              void* d_out, int out_size, void* d_ws, size_t ws_size,
                              hipStream_t stream) {
    const float* x   = (const float*)d_in[0];
    const float* W1  = (const float*)d_in[1];
    const float* b1  = (const float*)d_in[2];
    const float* W2  = (const float*)d_in[3];
    const float* b2  = (const float*)d_in[4];
    const float* Wm1 = (const float*)d_in[5];
    const float* bm1 = (const float*)d_in[6];
    const float* Wm2 = (const float*)d_in[7];
    const float* bm2 = (const float*)d_in[8];
    const int*   src = (const int*)d_in[9];
    const int*   dst = (const int*)d_in[10];

    const int N = in_sizes[0] / FEAT;
    const int E = in_sizes[9];

    // ---- workspace carve (256B aligned), with buffer reuse ----
    size_t off = 0;
    auto carve = [&](size_t bytes) -> char* {
        char* p = (char*)d_ws + off;
        off = (off + bytes + 255) & ~(size_t)255;
        return p;
    };
    float* norm_src = (float*)carve((size_t)N * 4);
    float* norm_dst = (float*)carve((size_t)N * 4);
    float* h_scaled = (float*)carve((size_t)N * FEAT * 4);   // also reused as MLP hidden
    float* agg      = (float*)carve((size_t)N * FEAT * 4);   // also reused as GEMM output
    bf16*  a_bf     = (bf16*) carve((size_t)N * FEAT * 2);
    bf16*  W1T      = (bf16*) carve((size_t)FEAT * FEAT * 2);
    bf16*  W2T      = (bf16*) carve((size_t)FEAT * FEAT * 2);
    bf16*  Wm1T     = (bf16*) carve((size_t)FEAT * MLPH * 2);
    float* hm       = h_scaled;                               // alias: free after layer2 agg

    const int TB = 256;
    const int nodeElemBlocks = (N * (FEAT / 4) + TB - 1) / TB;
    const int edgeBlocks     = (E + TB - 1) / TB;
    const int aggBlocks      = ((size_t)E * 64 + TB - 1) / TB;
    dim3 gemmGridH((N + 127) / 128, FEAT / 16);
    dim3 gemmGridM((N + 127) / 128, MLPH / 16);

    // ---- degree norms ----
    hipMemsetAsync(norm_src, 0, (size_t)N * 4, stream);
    hipMemsetAsync(norm_dst, 0, (size_t)N * 4, stream);
    gcn_deg_kernel<<<edgeBlocks, TB, 0, stream>>>(src, dst, norm_src, norm_dst, E);
    gcn_norm_kernel<<<(N + TB - 1) / TB, TB, 0, stream>>>(norm_src, norm_dst, N);

    // ---- weight prep (bf16, transposed) ----
    gcn_wprep_kernel<<<(FEAT * FEAT + TB - 1) / TB, TB, 0, stream>>>(W1, W1T, FEAT, FEAT);
    gcn_wprep_kernel<<<(FEAT * FEAT + TB - 1) / TB, TB, 0, stream>>>(W2, W2T, FEAT, FEAT);
    gcn_wprep_kernel<<<(FEAT * MLPH + TB - 1) / TB, TB, 0, stream>>>(Wm1, Wm1T, FEAT, MLPH);

    // ---- layer 1: scale -> scatter-add -> norm+bf16 -> WMMA GEMM(+ReLU) ----
    gcn_scale_kernel<<<nodeElemBlocks, TB, 0, stream>>>(x, norm_src, h_scaled, N);
    hipMemsetAsync(agg, 0, (size_t)N * FEAT * 4, stream);
    gcn_agg_kernel<<<aggBlocks, TB, 0, stream>>>(h_scaled, src, dst, agg, E);
    gcn_tobf16_kernel<<<nodeElemBlocks, TB, 0, stream>>>(agg, norm_dst, a_bf, N);
    gcn_gemm_wmma<FEAT><<<gemmGridH, TB, 0, stream>>>(a_bf, W1T, b1, agg, N, 1); // h1 -> agg

    // ---- layer 2 ----
    gcn_scale_kernel<<<nodeElemBlocks, TB, 0, stream>>>(agg, norm_src, h_scaled, N);
    hipMemsetAsync(agg, 0, (size_t)N * FEAT * 4, stream);
    gcn_agg_kernel<<<aggBlocks, TB, 0, stream>>>(h_scaled, src, dst, agg, E);
    gcn_tobf16_kernel<<<nodeElemBlocks, TB, 0, stream>>>(agg, norm_dst, a_bf, N);
    gcn_gemm_wmma<FEAT><<<gemmGridH, TB, 0, stream>>>(a_bf, W2T, b2, agg, N, 1); // h2 -> agg

    // ---- MLP head ----
    gcn_tobf16_kernel<<<nodeElemBlocks, TB, 0, stream>>>(agg, nullptr, a_bf, N);
    gcn_gemm_wmma<MLPH><<<gemmGridM, TB, 0, stream>>>(a_bf, Wm1T, bm1, hm, N, 1);
    gcn_mlp2_kernel<<<(N + TB - 1) / TB, TB, 0, stream>>>(hm, Wm2, bm2, (float*)d_out, N);
}
